// MoEModule_31705448579693
// MI455X (gfx1250) — compile-verified, matrix operand order checked by reading, not run on hardware
//
#include <hip/hip_runtime.h>
#include <hip/hip_bf16.h>

// ---------------- problem constants (from reference) ----------------
#define NTOK   4096              // B*T
#define DDIM   1024              // D
#define HDIM   4096              // 4*D
#define NEXP   8                 // E
#define TOPKK  2
#define MTILE  32                // token rows per FFN workgroup
#define MAX_SLOTS (NTOK*TOPKK + NEXP*MTILE)   // padded slot upper bound = 8448
#define MAX_TILES (MAX_SLOTS / MTILE)         // 264
// LDS: X and h tiles stored in WMMA-operand order ("swizzled"):
//   tile(kc, rb) = 32 lanes x 16 bf16, contiguous 1KB; A-load = tilebase+lane*32B
#define LDS_X_BYTES (MTILE*DDIM*2)            // 64 KB  = 32 kc x 2 rb x 1KB
#define LDS_H_BYTES (MTILE*HDIM*2)            // 256 KB = 128 kc x 2 rb x 1KB
#define LDS_BYTES   (LDS_X_BYTES + LDS_H_BYTES)  // 327680 = full 320KB WGP LDS

typedef __attribute__((ext_vector_type(16))) __bf16 bf16x16;
typedef __attribute__((ext_vector_type(8)))  __bf16 bf16x8;
typedef __attribute__((ext_vector_type(8)))  float  f32x8;

// ---------------- device helpers ----------------
// Branch-free exact-form gelu: 0.5*x*(1+erf(x/sqrt(2))) with Abramowitz-
// Stegun 7.1.26 erf (|abs err| <= 1.5e-7; no exec-mask branches, unlike
// libm erff which lowers to range-split code with s_and_saveexec).
__device__ __forceinline__ float gelu_exact(float v) {
    float y = v * 0.70710678118654752440f;
    float z = fabsf(y);
    float t = __frcp_rn(1.0f + 0.3275911f * z);
    float p = t * (0.254829592f +
              t * (-0.284496736f +
              t * (1.421413741f +
              t * (-1.453152027f +
              t * 1.061405429f))));
    float er = __builtin_fmaf(-p, __expf(-z * z), 1.0f);   // erf(|y|)
    er = __builtin_copysignf(er, y);
    return 0.5f * v * (1.0f + er);
}

// A-operand load from swizzled LDS: lane's 16 elements are contiguous 32B.
// Conflict-free: 32 lanes read 32B each, linearly across banks.
__device__ __forceinline__ bf16x16 load_a_sw(const __bf16* tilebase, int lane) {
    const __bf16* p = tilebase + lane * 16;
    bf16x8 lo = *(const bf16x8*)p;
    bf16x8 hi = *(const bf16x8*)(p + 8);
    bf16x16 v;
#pragma unroll
    for (int i = 0; i < 8; ++i) { v[i] = lo[i]; v[i + 8] = hi[i]; }
    return v;
}

// B-operand tile from fp32 weights in global memory: 16 floats per lane
// (4x b128) -> packed cvt to bf16 (weights stream once; no convert pass).
// rowp = W + (n0 + (lane&15)) * stride.
__device__ __forceinline__ bf16x16 load_b_f32(const float* rowp, int k0, int half) {
    const float* q = rowp + k0 + half * 8;
    float t[16];
    *(float4*)&t[0]  = *(const float4*)(q);
    *(float4*)&t[4]  = *(const float4*)(q + 4);
    *(float4*)&t[8]  = *(const float4*)(q + 16);
    *(float4*)&t[12] = *(const float4*)(q + 20);
    bf16x16 v;
#pragma unroll
    for (int i = 0; i < 16; ++i) v[i] = (__bf16)t[i];
    return v;
}

// ---------------- kernel 0: zero counters ----------------
__global__ void moe_init_kernel(int* counts, int* cursors) {
    int t = threadIdx.x;
    if (t < NEXP) { counts[t] = 0; cursors[t] = 0; }
}

// ---------------- kernel 1: router (logits -> softmax -> top2 -> gates) ----
__global__ void moe_router_kernel(const float* __restrict__ x,
                                  const float* __restrict__ rw,
                                  int* __restrict__ topk_idx,
                                  float* __restrict__ topk_w,
                                  int* __restrict__ counts) {
    int n = blockIdx.x * blockDim.x + threadIdx.x;
    if (n >= NTOK) return;
    const float* xr = x + (size_t)n * DDIM;
    float acc[NEXP];
#pragma unroll
    for (int e = 0; e < NEXP; ++e) acc[e] = 0.f;
    for (int d = 0; d < DDIM; d += 4) {
        float4 xv = *(const float4*)(xr + d);
#pragma unroll
        for (int e = 0; e < NEXP; ++e) {
            const float* r = rw + e * DDIM + d;
            acc[e] += xv.x * r[0] + xv.y * r[1] + xv.z * r[2] + xv.w * r[3];
        }
    }
    float mx = acc[0];
#pragma unroll
    for (int e = 1; e < NEXP; ++e) mx = fmaxf(mx, acc[e]);
    float p[NEXP];
#pragma unroll
    for (int e = 0; e < NEXP; ++e) p[e] = __expf(acc[e] - mx);
    int i0 = 0;
#pragma unroll
    for (int e = 1; e < NEXP; ++e) if (p[e] > p[i0]) i0 = e;
    int i1 = (i0 == 0) ? 1 : 0;
#pragma unroll
    for (int e = 0; e < NEXP; ++e) if (e != i0 && p[e] > p[i1]) i1 = e;
    float s = p[i0] + p[i1];                 // renormalize over top-2
    topk_idx[n * 2 + 0] = i0;  topk_w[n * 2 + 0] = p[i0] / s;
    topk_idx[n * 2 + 1] = i1;  topk_w[n * 2 + 1] = p[i1] / s;
    atomicAdd(&counts[i0], 1);
    atomicAdd(&counts[i1], 1);
}

// ---------------- kernel 2: build padded segments + tile table -------------
__global__ void moe_build_kernel(const int* __restrict__ counts,
                                 int* __restrict__ offsets,
                                 int* __restrict__ tile_expert,
                                 int* __restrict__ total_tiles,
                                 int* __restrict__ cursors,
                                 int* __restrict__ token_ids,
                                 float* __restrict__ gates) {
    if (threadIdx.x != 0 || blockIdx.x != 0) return;
    int off = 0;
    for (int e = 0; e < NEXP; ++e) {
        offsets[e] = off;
        int c = counts[e];
        int padded = (c + MTILE - 1) & ~(MTILE - 1);
        for (int t = off / MTILE; t < (off + padded) / MTILE; ++t) tile_expert[t] = e;
        for (int s = off + c; s < off + padded; ++s) { token_ids[s] = -1; gates[s] = 0.f; }
        cursors[e] = 0;
        off += padded;
    }
    offsets[NEXP] = off;
    *total_tiles = off / MTILE;
}

// ---------------- kernel 3: scatter tokens into expert slots ---------------
__global__ void moe_scatter_kernel(const int* __restrict__ topk_idx,
                                   const float* __restrict__ topk_w,
                                   const int* __restrict__ offsets,
                                   int* __restrict__ cursors,
                                   int* __restrict__ token_ids,
                                   float* __restrict__ gates) {
    int n = blockIdx.x * blockDim.x + threadIdx.x;
    if (n >= NTOK) return;
#pragma unroll
    for (int j = 0; j < TOPKK; ++j) {
        int e = topk_idx[n * 2 + j];
        int pos = atomicAdd(&cursors[e], 1);
        int s = offsets[e] + pos;
        token_ids[s] = n;
        gates[s] = topk_w[n * 2 + j];
    }
}

// ---------------- kernel 4: x fp32 -> bf16 (8 MB, L2-resident) -------------
__global__ void moe_xcast_kernel(const float* __restrict__ x, __bf16* __restrict__ xb) {
    size_t i = ((size_t)blockIdx.x * blockDim.x + threadIdx.x) * 8;
    float4 a = *(const float4*)(x + i);
    float4 b = *(const float4*)(x + i + 4);
    bf16x8 o;
    o[0] = (__bf16)a.x; o[1] = (__bf16)a.y; o[2] = (__bf16)a.z; o[3] = (__bf16)a.w;
    o[4] = (__bf16)b.x; o[5] = (__bf16)b.y; o[6] = (__bf16)b.z; o[7] = (__bf16)b.w;
    *(bf16x8*)(xb + i) = o;
}

// ---------------- kernel 5: zero output (harness poisons d_out) ------------
__global__ void moe_zero_kernel(float* __restrict__ out) {
    size_t i = ((size_t)blockIdx.x * blockDim.x + threadIdx.x) * 4;
    *(float4*)(out + i) = make_float4(0.f, 0.f, 0.f, 0.f);
}

// ---------------- kernel 6: fused expert FFN (WMMA bf16) -------------------
// One WG (8 wave32) per 32-token tile of one expert. X tile + full h tile
// live in 320KB LDS in WMMA-operand order (conflict-free A loads). Each wave
// processes 2 column-tiles per step: A tiles shared by 2 B tiles -> 4
// independent v_wmma_f32_16x16x32_bf16 chains, half the LDS read traffic.
__global__ void __launch_bounds__(256) moe_ffn_kernel(
        const __bf16* __restrict__ xb,
        const float* __restrict__ w1,     // [E][HDIM][DDIM]
        const float* __restrict__ w2,     // [E][DDIM][HDIM]
        const int* __restrict__ tile_expert,
        const int* __restrict__ total_tiles,
        const int* __restrict__ token_ids,
        const float* __restrict__ gates,
        float* __restrict__ out) {
    extern __shared__ __align__(16) char smem[];
    __bf16* lds_x = (__bf16*)smem;                    // 32 kc x 2 rb x 512 elem
    __bf16* lds_h = (__bf16*)(smem + LDS_X_BYTES);    // 128 kc x 2 rb x 512 elem

    int tile = blockIdx.x;
    if (tile >= *total_tiles) return;
    const int e = tile_expert[tile];
    const int slot_base = tile * MTILE;

    const int tid  = threadIdx.x;
    const int wave = tid >> 5;
    const int lane = tid & 31;
    const int half = lane >> 4;
    const int rr   = lane & 15;

    // ---- stage X tile (gathered rows, pad rows zeroed) into swizzled LDS ----
    // element (row r, col c) -> tile(kc=c/32, rb=r/16), lane'=(r&15)+16*((c>>3)&1),
    // idx = (c&7) + ((c&16)?8:0); an 8-aligned 16B chunk stays one uint4 store.
#pragma unroll
    for (int it = 0; it < 16; ++it) {
        int chunk = it * 256 + tid;            // 4096 chunks of 8 bf16
        int r = chunk >> 7;                    // 128 chunks per row
        int c = (chunk & 127) * 8;
        int tok = token_ids[slot_base + r];
        uint4 v = make_uint4(0u, 0u, 0u, 0u);
        if (tok >= 0) v = *(const uint4*)(xb + (size_t)tok * DDIM + c);
        int kc = c >> 5, c8 = c & 31;
        int lane_p = (r & 15) + (((c8 >> 3) & 1) << 4);
        int idx0 = (c8 & 16) ? 8 : 0;
        *(uint4*)(lds_x + ((kc * 2 + (r >> 4)) << 9) + lane_p * 16 + idx0) = v;
    }
    __syncthreads();

    // ---- Phase A: h = gelu(X @ w1[e]^T); wave w owns cols [w*512, w*512+512)
    const float* W1 = w1 + (size_t)e * HDIM * DDIM;
    const int lane_hi = (rr >> 3) << 4;   // lane' high part for h stores
    const int idx_lo  = rr & 7;
    for (int ct = 0; ct < 16; ++ct) {
        const int f0 = wave * 512 + ct * 32;            // two 16-col subtiles
        const float* w1rowA = W1 + (size_t)(f0 + rr) * DDIM;
        const float* w1rowB = W1 + (size_t)(f0 + 16 + rr) * DDIM;
        __builtin_prefetch(w1rowA + (size_t)DDIM * 32, 0, 0);  // next ct rows
        f32x8 acc00 = {}, acc01 = {}, acc10 = {}, acc11 = {};
#pragma unroll 2
        for (int k0 = 0; k0 < DDIM; k0 += 32) {
            const int kc = k0 >> 5;
            bf16x16 a0 = load_a_sw(lds_x + ((kc * 2 + 0) << 9), lane);
            bf16x16 a1 = load_a_sw(lds_x + ((kc * 2 + 1) << 9), lane);
            bf16x16 b0 = load_b_f32(w1rowA, k0, half);
            bf16x16 b1 = load_b_f32(w1rowB, k0, half);
            acc00 = __builtin_amdgcn_wmma_f32_16x16x32_bf16(false, a0, false, b0,
                                                            (short)0, acc00, false, false);
            acc01 = __builtin_amdgcn_wmma_f32_16x16x32_bf16(false, a1, false, b0,
                                                            (short)0, acc01, false, false);
            acc10 = __builtin_amdgcn_wmma_f32_16x16x32_bf16(false, a0, false, b1,
                                                            (short)0, acc10, false, false);
            acc11 = __builtin_amdgcn_wmma_f32_16x16x32_bf16(false, a1, false, b1,
                                                            (short)0, acc11, false, false);
        }
        // gelu + store into swizzled h. D layout: VGPR i, lane -> row
        // m = half*8+i (+16 for acc*1), col = f0(+16) + rr.
        const int kc_h = f0 >> 5;
#pragma unroll
        for (int i = 0; i < 8; ++i) {
            const int lp = (half * 8 + i) + lane_hi;
            __bf16* t0 = lds_h + ((kc_h * 2 + 0) << 9) + lp * 16;  // rows 0-15
            __bf16* t1 = lds_h + ((kc_h * 2 + 1) << 9) + lp * 16;  // rows 16-31
            t0[idx_lo]     = (__bf16)gelu_exact(acc00[i]);
            t0[idx_lo + 8] = (__bf16)gelu_exact(acc10[i]);
            t1[idx_lo]     = (__bf16)gelu_exact(acc01[i]);
            t1[idx_lo + 8] = (__bf16)gelu_exact(acc11[i]);
        }
    }
    __syncthreads();

    // ---- preload tokens/gates for the tile (invariant over output tiles) ----
    int   tk0[8], tk1[8];
    float g0[8],  g1[8];
#pragma unroll
    for (int i = 0; i < 8; ++i) {
        int m0 = half * 8 + i, m1 = m0 + 16;
        tk0[i] = token_ids[slot_base + m0];  g0[i] = gates[slot_base + m0];
        tk1[i] = token_ids[slot_base + m1];  g1[i] = gates[slot_base + m1];
    }

    // ---- Phase B: y = h @ w2[e]^T; wave w owns out cols [w*128, w*128+128) ----
    const float* W2 = w2 + (size_t)e * DDIM * HDIM;
    for (int ct = 0; ct < 4; ++ct) {
        const int d0 = wave * 128 + ct * 32;
        const float* w2rowA = W2 + (size_t)(d0 + rr) * HDIM;
        const float* w2rowB = W2 + (size_t)(d0 + 16 + rr) * HDIM;
        __builtin_prefetch(w2rowA + (size_t)HDIM * 32, 0, 0);
        f32x8 acc00 = {}, acc01 = {}, acc10 = {}, acc11 = {};
#pragma unroll 2
        for (int k0 = 0; k0 < HDIM; k0 += 32) {
            const int kc = k0 >> 5;
            bf16x16 a0 = load_a_sw(lds_h + ((kc * 2 + 0) << 9), lane);
            bf16x16 a1 = load_a_sw(lds_h + ((kc * 2 + 1) << 9), lane);
            bf16x16 b0 = load_b_f32(w2rowA, k0, half);
            bf16x16 b1 = load_b_f32(w2rowB, k0, half);
            acc00 = __builtin_amdgcn_wmma_f32_16x16x32_bf16(false, a0, false, b0,
                                                            (short)0, acc00, false, false);
            acc01 = __builtin_amdgcn_wmma_f32_16x16x32_bf16(false, a1, false, b0,
                                                            (short)0, acc01, false, false);
            acc10 = __builtin_amdgcn_wmma_f32_16x16x32_bf16(false, a0, false, b1,
                                                            (short)0, acc10, false, false);
            acc11 = __builtin_amdgcn_wmma_f32_16x16x32_bf16(false, a1, false, b1,
                                                            (short)0, acc11, false, false);
        }
        const int colA = d0 + rr, colB = d0 + 16 + rr;
#pragma unroll
        for (int i = 0; i < 8; ++i) {
            if (tk0[i] >= 0) {
                atomicAdd(out + (size_t)tk0[i] * DDIM + colA, g0[i] * acc00[i]);
                atomicAdd(out + (size_t)tk0[i] * DDIM + colB, g0[i] * acc10[i]);
            }
            if (tk1[i] >= 0) {
                atomicAdd(out + (size_t)tk1[i] * DDIM + colA, g1[i] * acc01[i]);
                atomicAdd(out + (size_t)tk1[i] * DDIM + colB, g1[i] * acc11[i]);
            }
        }
    }
}

// ---------------- host side ----------------
extern "C" void kernel_launch(void* const* d_in, const int* in_sizes, int n_in,
                              void* d_out, int out_size, void* d_ws, size_t ws_size,
                              hipStream_t stream) {
    const float* x  = (const float*)d_in[0];   // [4096,1024]
    const float* rw = (const float*)d_in[1];   // [8,1024]
    const float* w1 = (const float*)d_in[2];   // [8,4096,1024]
    const float* w2 = (const float*)d_in[3];   // [8,1024,4096]
    float* out = (float*)d_out;
    (void)in_sizes; (void)n_in; (void)out_size;

    // workspace layout
    char* ws = (char*)d_ws;
    size_t o = 0;
    auto alloc = [&](size_t bytes, size_t align) -> char* {
        o = (o + align - 1) & ~(align - 1);
        char* p = ws + o; o += bytes; return p;
    };
    int*    counts      = (int*)alloc(NEXP * 4, 256);
    int*    cursors     = (int*)alloc(NEXP * 4, 256);
    int*    offsets     = (int*)alloc((NEXP + 1) * 4, 256);
    int*    total_tiles = (int*)alloc(4, 256);
    int*    tile_expert = (int*)alloc(MAX_TILES * 4, 256);
    int*    topk_idx    = (int*)alloc((size_t)NTOK * 2 * 4, 256);
    float*  topk_w      = (float*)alloc((size_t)NTOK * 2 * 4, 256);
    int*    token_ids   = (int*)alloc((size_t)MAX_SLOTS * 4, 256);
    float*  gates       = (float*)alloc((size_t)MAX_SLOTS * 4, 256);
    __bf16* xb          = (__bf16*)alloc((size_t)NTOK * DDIM * 2, 256);
    (void)ws_size;

    // allow full-WGP dynamic LDS for the FFN kernel (idempotent, capture-safe,
    // called unconditionally every launch -- no static guards)
    (void)hipFuncSetAttribute((const void*)moe_ffn_kernel,
                              hipFuncAttributeMaxDynamicSharedMemorySize,
                              LDS_BYTES);

    moe_init_kernel<<<1, 64, 0, stream>>>(counts, cursors);
    moe_router_kernel<<<NTOK / 128, 128, 0, stream>>>(x, rw, topk_idx, topk_w, counts);
    moe_build_kernel<<<1, 1, 0, stream>>>(counts, offsets, tile_expert, total_tiles,
                                          cursors, token_ids, gates);
    moe_scatter_kernel<<<NTOK / 256, 256, 0, stream>>>(topk_idx, topk_w, offsets,
                                                       cursors, token_ids, gates);
    moe_xcast_kernel<<<(NTOK * DDIM) / (256 * 8), 256, 0, stream>>>(x, xb);
    moe_zero_kernel<<<(NTOK * DDIM) / (256 * 4), 256, 0, stream>>>(out);
    moe_ffn_kernel<<<MAX_TILES, 256, LDS_BYTES, stream>>>(
        xb, w1, w2, tile_expert, total_tiles, token_ids, gates, out);
}
